// FeedForward_32469952758514
// MI455X (gfx1250) — matrix operand order repaired and checked by reading, not compiled, not run on hardware
//
#include <hip/hip_runtime.h>
#include <stdint.h>

// Problem constants (match reference)
#define H_DIM   2048
#define I_DIM   1408
#define E_NUM   16
#define G_NUM   4
#define TOPKG   2
#define K_TOP   4
#define SI_DIM  2816          // 2*I (shared-expert intermediate)
#define T_TOK   2048          // B*S

typedef __bf16    bf16_t;
typedef __attribute__((ext_vector_type(16))) __bf16    v16bf;
typedef __attribute__((ext_vector_type(8)))  float     v8f;
typedef __attribute__((ext_vector_type(4)))  uint32_t  v4u;

union BV { v4u q[2]; v16bf v; };   // 32B: two b128 loads -> one WMMA 16-bf16 operand

// ---- fp32 -> bf16 (round to nearest even) ----
__device__ __forceinline__ uint16_t f2bf(float f) {
  uint32_t u = __float_as_uint(f);
  u += 0x7FFFu + ((u >> 16) & 1u);
  return (uint16_t)(u >> 16);
}
__device__ __forceinline__ uint32_t pack2bf(float a, float b) {
  return (uint32_t)f2bf(a) | ((uint32_t)f2bf(b) << 16);
}

// ================= Stage 1: bulk fp32 -> bf16 conversion =================
__global__ void cvt_f32_bf16(const float4* __restrict__ src,
                             uint2* __restrict__ dst, long long n4) {
  long long i = (long long)blockIdx.x * blockDim.x + threadIdx.x;
  long long stride = (long long)gridDim.x * blockDim.x;
  for (; i < n4; i += stride) {
    float4 v = src[i];
    uint2 o;
    o.x = pack2bf(v.x, v.y);
    o.y = pack2bf(v.z, v.w);
    dst[i] = o;
  }
}

// ================= Stage 2: fp32 gating (group-limited top-k) =================
__global__ void gate_kernel(const float* __restrict__ x, const float* __restrict__ gw,
                            float* __restrict__ topw, int* __restrict__ topid) {
  int t = blockIdx.x;
  __shared__ float part[128];
  __shared__ float sc[E_NUM];
  int tid = threadIdx.x;
  int e = tid >> 3, p = tid & 7;
  const float* xr = x + (size_t)t * H_DIM;
  const float* gr = gw + (size_t)e * H_DIM;
  float s = 0.f;
  int k0 = p * (H_DIM / 8);
  for (int k = 0; k < H_DIM / 8; ++k) s += xr[k0 + k] * gr[k0 + k];
  part[tid] = s;
  __syncthreads();
  if (p == 0) {
    float tot = 0.f;
    for (int j = 0; j < 8; ++j) tot += part[e * 8 + j];
    sc[e] = tot;
  }
  __syncthreads();
  if (tid == 0) {
    float mx = sc[0];
    for (int i = 1; i < E_NUM; ++i) mx = fmaxf(mx, sc[i]);
    float pr[E_NUM]; float sum = 0.f;
    for (int i = 0; i < E_NUM; ++i) { pr[i] = expf(sc[i] - mx); sum += pr[i]; }
    for (int i = 0; i < E_NUM; ++i) pr[i] /= sum;
    float gs[G_NUM];
    for (int g = 0; g < G_NUM; ++g) {
      float m = pr[g * 4];
      for (int j = 1; j < 4; ++j) m = fmaxf(m, pr[g * 4 + j]);
      gs[g] = m;
    }
    int g1 = 0;
    for (int g = 1; g < G_NUM; ++g) if (gs[g] > gs[g1]) g1 = g;
    int g2 = -1;
    for (int g = 0; g < G_NUM; ++g)
      if (g != g1 && (g2 < 0 || gs[g] > gs[g2])) g2 = g;
    float msk[E_NUM];
    for (int i = 0; i < E_NUM; ++i) {
      int g = i >> 2;
      msk[i] = (g == g1 || g == g2) ? pr[i] : -1.f;
    }
    int ids[K_TOP]; float wv[K_TOP]; float wsum = 0.f;
    for (int k = 0; k < K_TOP; ++k) {
      int best = 0;
      for (int i = 1; i < E_NUM; ++i) if (msk[i] > msk[best]) best = i;
      ids[k] = best; wv[k] = msk[best]; wsum += msk[best];
      msk[best] = -2.f;
    }
    wsum += 1e-20f;
    for (int k = 0; k < K_TOP; ++k) {
      topw[t * K_TOP + k] = wv[k] / wsum;   // SCALE = 1.0
      topid[t * K_TOP + k] = ids[k];
    }
  }
}

// ================= Stage 3: deterministic per-expert token lists =================
__global__ void build_lists(const int* __restrict__ topid,
                            int* __restrict__ lists, int* __restrict__ counts) {
  int e = blockIdx.x;
  if (threadIdx.x != 0) return;
  int n = 0;
  for (int t = 0; t < T_TOK; ++t)
    for (int s = 0; s < K_TOP; ++s)
      if (topid[t * K_TOP + s] == e) lists[e * T_TOK + n++] = (t << 2) | s;
  counts[e] = n;
}

// ================= Stage 4/5: fused up -> SwiGLU -> down, bf16 WMMA =================
// IC = act (intermediate) width: 1408 routed, 2816 shared. Up weight is [2*IC, H],
// down weight row length is IC. 16-token row tile per block, 256 threads (8 waves).
template <int IC, bool ROUTED>
__global__ __launch_bounds__(256)
void moe_ffn(const uint16_t* __restrict__ xb,     // [T][H] bf16
             const uint16_t* __restrict__ upw,    // [(E)][2*IC][H] bf16
             const uint16_t* __restrict__ dnw,    // [(E)][H][IC] bf16
             const int* __restrict__ lists, const int* __restrict__ counts,
             const float* __restrict__ topw,
             float* __restrict__ out)              // [T][H] f32
{
  constexpr int XPITCH = H_DIM + 8;   // bf16 elems, bank-stagger
  constexpr int APITCH = IC + 8;
  extern __shared__ char smem[];
  uint16_t* xt  = (uint16_t*)smem;                                   // 16 x XPITCH
  uint16_t* act = (uint16_t*)(smem + 16 * XPITCH * 2);               // 16 x APITCH
  float*    sW  = (float*)(smem + 16 * XPITCH * 2 + 16 * APITCH * 2);
  int*      sTok = (int*)(sW + 16);

  int e, tile;
  if constexpr (ROUTED) {
    e = blockIdx.x >> 7;                 // / (T_TOK/16)
    tile = blockIdx.x & 127;
    if (tile * 16 >= counts[e]) return;  // block-uniform early exit
  } else {
    e = 0; tile = blockIdx.x;
  }

  int tid = threadIdx.x;
  if (tid < 16) {
    if constexpr (ROUTED) {
      int idx = tile * 16 + tid;
      if (idx < counts[e]) {
        int pk = lists[e * T_TOK + idx];
        sTok[tid] = pk >> 2;
        sW[tid] = topw[(pk >> 2) * K_TOP + (pk & 3)];
      } else { sTok[tid] = 0; sW[tid] = 0.f; }
    } else {
      sTok[tid] = tile * 16 + tid; sW[tid] = 1.f;
    }
  }
  __syncthreads();

  // Gather 16 token rows (bf16) into LDS, 16B chunks.
  for (int i = tid; i < 16 * (H_DIM / 8); i += 256) {
    int r = i >> 8, c = i & 255;
    const v4u* src = (const v4u*)(xb + (size_t)sTok[r] * H_DIM) + c;
    *(v4u*)((char*)xt + (size_t)r * XPITCH * 2 + c * 16) = *src;
  }
  __syncthreads();

  const int lane = tid & 31, wv = tid >> 5;
  const int mrow = lane & 15, khalf = lane >> 4;
  const uint16_t* upE = upw + (ROUTED ? (size_t)e * 2 * IC * H_DIM : 0);
  const uint16_t* dnE = dnw + (ROUTED ? (size_t)e * H_DIM * IC : 0);

  // ---- UP + SwiGLU: each wave owns IC/128 act tiles; G and U accum in-regs ----
  constexpr int ATPW = IC / 128;   // act tiles per wave (11 or 22)
  for (int ti = 0; ti < ATPW; ++ti) {
    int c0 = (wv * ATPW + ti) * 16;
    v8f accG = {}; v8f accU = {};
    const uint16_t* gRow = upE + (size_t)(c0 + mrow) * H_DIM;
    const uint16_t* uRow = upE + (size_t)(c0 + mrow + IC) * H_DIM;
    const char* arow = (const char*)xt + (size_t)mrow * XPITCH * 2;
    for (int k0 = 0; k0 < H_DIM; k0 += 32) {
      BV a, bg, bu;
      int ak = k0 + khalf * 8;                        // A: K split across lane halves
      a.q[0] = *(const v4u*)(arow + ak * 2);
      a.q[1] = *(const v4u*)(arow + ak * 2 + 32);
      int bk = k0 + khalf * 8;                        // B: K split across lane halves (same as A)
      bg.q[0] = *(const v4u*)(gRow + bk);
      bg.q[1] = *(const v4u*)(gRow + bk + 16);
      bu.q[0] = *(const v4u*)(uRow + bk);
      bu.q[1] = *(const v4u*)(uRow + bk + 16);
      accG = __builtin_amdgcn_wmma_f32_16x16x32_bf16(false, a.v, false, bg.v,
                                                     (short)0, accG, false, false);
      accU = __builtin_amdgcn_wmma_f32_16x16x32_bf16(false, a.v, false, bu.v,
                                                     (short)0, accU, false, false);
    }
#pragma unroll
    for (int v = 0; v < 8; ++v) {       // SwiGLU, register-paired G/U
      float g = accG[v], u = accU[v];
      float sact = (g / (1.f + expf(-g))) * u;
      int m = v + khalf * 8;
      *(uint16_t*)((char*)act + (size_t)m * APITCH * 2 + (size_t)(c0 + mrow) * 2) =
          f2bf(sact);
    }
  }
  __syncthreads();

  // ---- DOWN: each wave owns 16 output col tiles (2048/16/8) ----
  for (int ot = 0; ot < 16; ++ot) {
    int n0 = (wv * 16 + ot) * 16;
    v8f acc = {};
    const uint16_t* dRow = dnE + (size_t)(n0 + mrow) * IC;
    const char* arow = (const char*)act + (size_t)mrow * APITCH * 2;
    for (int k0 = 0; k0 < IC; k0 += 32) {
      BV a, b;
      int ak = k0 + khalf * 8;
      a.q[0] = *(const v4u*)(arow + ak * 2);
      a.q[1] = *(const v4u*)(arow + ak * 2 + 32);
      int bk = k0 + khalf * 8;
      b.q[0] = *(const v4u*)(dRow + bk);
      b.q[1] = *(const v4u*)(dRow + bk + 16);
      acc = __builtin_amdgcn_wmma_f32_16x16x32_bf16(false, a.v, false, b.v,
                                                    (short)0, acc, false, false);
    }
#pragma unroll
    for (int v = 0; v < 8; ++v) {
      int m = v + khalf * 8;
      size_t oidx = (size_t)sTok[m] * H_DIM + n0 + mrow;
      if constexpr (ROUTED) {
        float w = sW[m];
        if (w != 0.f) atomicAdd(&out[oidx], w * acc[v]);
      } else {
        out[oidx] = acc[v];           // shared expert initializes out
      }
    }
  }
}

// ================= host-side launcher =================
extern "C" void kernel_launch(void* const* d_in, const int* in_sizes, int n_in,
                              void* d_out, int out_size, void* d_ws, size_t ws_size,
                              hipStream_t stream) {
  (void)in_sizes; (void)n_in; (void)out_size; (void)ws_size;
  const float* x   = (const float*)d_in[0];
  const float* gw  = (const float*)d_in[1];
  const float* upw = (const float*)d_in[2];
  const float* dnw = (const float*)d_in[3];
  const float* sup = (const float*)d_in[4];
  const float* sdn = (const float*)d_in[5];
  float* out = (float*)d_out;

  // Workspace carve-up (~320 MB bf16 weights + gating scratch).
  char* ws = (char*)d_ws;
  size_t off = 0;
  auto carve = [&](size_t bytes) {
    void* p = ws + off;
    off = (off + bytes + 255) & ~(size_t)255;
    return p;
  };
  uint16_t* xb   = (uint16_t*)carve((size_t)T_TOK * H_DIM * 2);
  uint16_t* upb  = (uint16_t*)carve((size_t)E_NUM * 2 * I_DIM * H_DIM * 2);
  uint16_t* dnb  = (uint16_t*)carve((size_t)E_NUM * H_DIM * I_DIM * 2);
  uint16_t* supb = (uint16_t*)carve((size_t)2 * SI_DIM * H_DIM * 2);
  uint16_t* sdnb = (uint16_t*)carve((size_t)H_DIM * SI_DIM * 2);
  float*    topw = (float*)carve((size_t)T_TOK * K_TOP * 4);
  int*      topid = (int*)carve((size_t)T_TOK * K_TOP * 4);
  int*      lists = (int*)carve((size_t)E_NUM * T_TOK * 4);
  int*      counts = (int*)carve((size_t)E_NUM * 4);

  auto cvt = [&](const float* s, uint16_t* d, size_t n) {
    long long n4 = (long long)(n / 4);
    int blocks = (int)((n4 + 255) / 256);
    if (blocks > 16384) blocks = 16384;
    cvt_f32_bf16<<<blocks, 256, 0, stream>>>((const float4*)s, (uint2*)d, n4);
  };
  cvt(x,   xb,   (size_t)T_TOK * H_DIM);
  cvt(upw, upb,  (size_t)E_NUM * 2 * I_DIM * H_DIM);
  cvt(dnw, dnb,  (size_t)E_NUM * H_DIM * I_DIM);
  cvt(sup, supb, (size_t)2 * SI_DIM * H_DIM);
  cvt(sdn, sdnb, (size_t)H_DIM * SI_DIM);

  gate_kernel<<<T_TOK, 128, 0, stream>>>(x, gw, topw, topid);
  build_lists<<<E_NUM, 32, 0, stream>>>(topid, lists, counts);

  // LDS: x tile + act tile (+ tiny token/weight staging)
  const size_t smemS = (size_t)16 * (H_DIM + 8) * 2 + (size_t)16 * (SI_DIM + 8) * 2 + 128;
  const size_t smemR = (size_t)16 * (H_DIM + 8) * 2 + (size_t)16 * (I_DIM + 8) * 2 + 128;
  (void)hipFuncSetAttribute((const void*)&moe_ffn<SI_DIM, false>,
                            hipFuncAttributeMaxDynamicSharedMemorySize, (int)smemS);
  (void)hipFuncSetAttribute((const void*)&moe_ffn<I_DIM, true>,
                            hipFuncAttributeMaxDynamicSharedMemorySize, (int)smemR);

  // Shared expert first: plain stores initialize out.
  moe_ffn<SI_DIM, false><<<T_TOK / 16, 256, smemS, stream>>>(
      xb, supb, sdnb, nullptr, nullptr, nullptr, out);
  // Routed experts: atomic f32 accumulation scaled by combine weights.
  moe_ffn<I_DIM, true><<<E_NUM * (T_TOK / 16), 256, smemR, stream>>>(
      xb, upb, dnb, lists, counts, topw, out);
}
